// BEV_DFA_51376398795124
// MI455X (gfx1250) — matrix operand order repaired and checked by reading, not compiled, not run on hardware
//
#include <hip/hip_runtime.h>
#include <hip/hip_bf16.h>

// ---------------------------------------------------------------------------
// BEV deformable-attention stack for MI455X (gfx1250, wave32, WMMA).
// GEMMs: bf16 V_WMMA_F32_16X16X32_BF16, fp32 accum. Triple-buffered LDS fed
// by GLOBAL_LOAD_ASYNC_TO_LDS_B128 (ASYNCcnt); fully unrolled K pipeline with
// one s_barrier per K-step and 2 DMA stages in flight.
// ---------------------------------------------------------------------------

typedef __bf16 bf16_t;
typedef __attribute__((ext_vector_type(16))) __bf16 v16bf;
typedef __attribute__((ext_vector_type(8)))  float  v8f;

#define HEADS  8
#define POINTS 4
#define CDIM   256
#define BSZ    2
#define HH     200
#define WWD    200
#define HWSZ   (HH * WWD)          // 40000
#define MTOT   (BSZ * HWSZ)        // 80000

union Frag16 {
    v16bf v;
    uint4 u[2];
};

// Async memory->LDS copy, 16B per lane. ldsOff = LDS byte offset (low 32 bits
// of the generic pointer == offset from LDS_BASE), gAddr = 64-bit global addr.
#define ASYNC_LDS_CP_B128(ldsOff, gAddr)                                   \
    asm volatile("global_load_async_to_lds_b128 %0, %1, off"               \
                 :: "v"(ldsOff), "v"(gAddr) : "memory")

// ---------------------------------------------------------------------------
// Generic bf16 WMMA GEMM:  C[M,N] = A[M,K] @ B[N,K]^T + bias[N]
// Block tile 128x128, K-step 32, 8 waves arranged 4(M) x 2(N); each wave owns
// a 32x64 tile = 2x4 wmma accumulators. Triple-buffered LDS tiles filled by
// async DMA; compile-time K so the software pipeline fully unrolls.
// Caller guarantees M % 128 == 0 and B has >=128 readable rows per N-block
// (short-N weights padded + zero-filled), so tile loads are unguarded.
// ---------------------------------------------------------------------------
template <int K, bool RELU, bool STF32, bool STBF16>
__global__ __launch_bounds__(256)
void gemm_bf16_wmma(const bf16_t* __restrict__ A,
                    const bf16_t* __restrict__ B,
                    const float*  __restrict__ bias,
                    float*        __restrict__ Cf,
                    bf16_t*       __restrict__ Cb,
                    int M, int N)
{
    constexpr int BM = 128, BN = 128, BK = 32;
    constexpr int NK = K / BK;                      // 8 for K=256
    constexpr int LDS_STR = BK + 8;                 // 40 elems = 80B row (conflict-free)
    __shared__ bf16_t As[3][BM * LDS_STR];          // 3 x 10240 B
    __shared__ bf16_t Bs[3][BN * LDS_STR];          // 3 x 10240 B

    const int tid  = threadIdx.x;
    const int lane = tid & 31;
    const int wave = tid >> 5;
    const int mw   = wave >> 1;                     // 0..3
    const int nw   = wave & 1;                      // 0..1
    const int rowW = mw * 32;
    const int colW = nw * 64;
    const int l15  = lane & 15;
    const int lh   = lane >> 4;

    const int mBase = blockIdx.y * BM;
    const int nBase = blockIdx.x * BN;

    // Per-thread tile-fill coordinates: 512 16B vectors per tile, 2 per thread
    // for A and 2 for B. Vector vi covers tile row (vi>>2), k-chunk (vi&3)*8.
    const int rr = tid >> 2;                        // 0..63
    const int cv = (tid & 3) * 8;                   // 0,8,16,24
    const bf16_t* gA0 = A + (size_t)(mBase + rr)      * K + cv;
    const bf16_t* gA1 = A + (size_t)(mBase + rr + 64) * K + cv;
    const bf16_t* gB0 = B + (size_t)(nBase + rr)      * K + cv;
    const bf16_t* gB1 = B + (size_t)(nBase + rr + 64) * K + cv;
    const unsigned lA0 = (unsigned)(uintptr_t)&As[0][rr * LDS_STR + cv];
    const unsigned lA1 = (unsigned)(uintptr_t)&As[0][(rr + 64) * LDS_STR + cv];
    const unsigned lB0 = (unsigned)(uintptr_t)&Bs[0][rr * LDS_STR + cv];
    const unsigned lB1 = (unsigned)(uintptr_t)&Bs[0][(rr + 64) * LDS_STR + cv];
    constexpr unsigned BUFSTRIDE = (unsigned)(BM * LDS_STR * sizeof(bf16_t)); // 10240

    auto issue = [&](int k0, int buf) {
        const unsigned d = (unsigned)buf * BUFSTRIDE;
        ASYNC_LDS_CP_B128(lA0 + d, (unsigned long long)(uintptr_t)(gA0 + k0));
        ASYNC_LDS_CP_B128(lA1 + d, (unsigned long long)(uintptr_t)(gA1 + k0));
        ASYNC_LDS_CP_B128(lB0 + d, (unsigned long long)(uintptr_t)(gB0 + k0));
        ASYNC_LDS_CP_B128(lB1 + d, (unsigned long long)(uintptr_t)(gB1 + k0));
    };

    v8f acc[2][4];
#pragma unroll
    for (int i = 0; i < 2; ++i)
#pragma unroll
        for (int j = 0; j < 4; ++j)
#pragma unroll
            for (int r = 0; r < 8; ++r) acc[i][j][r] = 0.0f;

    // Prime 2 pipeline stages (8 async copies outstanding per thread).
    issue(0, 0);
    if (NK > 1) issue(BK, 1);

#pragma unroll
    for (int k = 0; k < NK; ++k) {
        const int buf = k % 3;
        // Retire this wave's stage-k copies (stage k+1 may remain in flight).
        if (k < NK - 1) asm volatile("s_wait_asynccnt 0x4" ::: "memory");
        else            asm volatile("s_wait_asynccnt 0x0" ::: "memory");
        __syncthreads();   // all waves' stage-k DMA visible; also proves all
                           // reads of buffer (k-1)%3 == (k+2)%3 are done.
        if (k + 2 < NK) issue((k + 2) * BK, (k + 2) % 3);

        // Fragments per ISA 7.12.2:
        // A (16x32): lane l15 -> row, elems 0..7 = K(lh*8..), 8..15 = K(16+lh*8..)
        Frag16 fa[2], fb[4];
#pragma unroll
        for (int sm = 0; sm < 2; ++sm) {
            const bf16_t* pa = &As[buf][(rowW + sm * 16 + l15) * LDS_STR + lh * 8];
            fa[sm].u[0] = *(const uint4*)pa;
            fa[sm].u[1] = *(const uint4*)(pa + 16);
        }
        // B (32x16): N = l15, elems 0..7 = K(lh*8..), 8..15 = K(16+lh*8..)
#pragma unroll
        for (int sn = 0; sn < 4; ++sn) {
            const bf16_t* pb = &Bs[buf][(colW + sn * 16 + l15) * LDS_STR + lh * 8];
            fb[sn].u[0] = *(const uint4*)pb;
            fb[sn].u[1] = *(const uint4*)(pb + 16);
        }

#pragma unroll
        for (int sm = 0; sm < 2; ++sm)
#pragma unroll
            for (int sn = 0; sn < 4; ++sn)
                acc[sm][sn] = __builtin_amdgcn_wmma_f32_16x16x32_bf16(
                    false, fa[sm].v, false, fb[sn].v,
                    (short)0, acc[sm][sn], false, false);
    }

    // Epilogue: bias (+ReLU) and stores. C/D layout: VGPR r -> row lh*8 + r, col = l15.
#pragma unroll
    for (int sn = 0; sn < 4; ++sn) {
        int col  = nBase + colW + sn * 16 + l15;
        float bb = (col < N) ? bias[col] : 0.0f;
#pragma unroll
        for (int sm = 0; sm < 2; ++sm) {
#pragma unroll
            for (int r = 0; r < 8; ++r) {
                int row = mBase + rowW + sm * 16 + lh * 8 + r;
                if (col < N) {
                    float v = acc[sm][sn][r] + bb;
                    if (RELU) v = fmaxf(v, 0.0f);
                    if (STF32)  Cf[(size_t)row * N + col] = v;
                    if (STBF16) Cb[(size_t)row * N + col] = (bf16_t)v;
                }
            }
        }
    }
}

// ---------------------------------------------------------------------------
// Deformable sampling + softmax attention. One wave per (b,q,head); 32 lanes
// cover head_dim=32. value (82MB fp32) is L2-resident (192MB L2).
// Coordinates simplify exactly: x = qx + off_x, y = qy + off_y.
// ---------------------------------------------------------------------------
__global__ __launch_bounds__(256)
void dfa_sample(const float* __restrict__ value,   // [M,256] = [b,q][h*32+d]
                const float* __restrict__ off,     // [M,64]
                const float* __restrict__ aw,      // [M,32]
                bf16_t*      __restrict__ attn)    // [M,256]
{
    const int bq   = blockIdx.x;                   // 0..MTOT-1
    const int h    = threadIdx.x >> 5;             // wave id = head
    const int lane = threadIdx.x & 31;             // channel within head
    const int b    = bq / HWSZ;
    const int q    = bq - b * HWSZ;
    const int qx   = q % WWD;
    const int qy   = q / WWD;

    const float* offp = off + (size_t)bq * 64 + h * 8;
    const float* awp  = aw  + (size_t)bq * 32 + h * 4;
    float a0 = awp[0], a1 = awp[1], a2 = awp[2], a3 = awp[3];
    float mx = fmaxf(fmaxf(a0, a1), fmaxf(a2, a3));
    float e0 = __expf(a0 - mx), e1 = __expf(a1 - mx);
    float e2 = __expf(a2 - mx), e3 = __expf(a3 - mx);
    float inv = 1.0f / (e0 + e1 + e2 + e3);
    float w[POINTS] = { e0 * inv, e1 * inv, e2 * inv, e3 * inv };

    const float* vbase = value + (size_t)b * HWSZ * CDIM + h * 32 + lane;
    float acc = 0.0f;
#pragma unroll
    for (int p = 0; p < POINTS; ++p) {
        float x   = (float)qx + offp[p * 2 + 0];
        float y   = (float)qy + offp[p * 2 + 1];
        float x0f = floorf(x), y0f = floorf(y);
        int   x0  = (int)x0f,  y0  = (int)y0f;
        float wx  = x - x0f,   wy  = y - y0f;
        float s = 0.0f;
#pragma unroll
        for (int cy = 0; cy < 2; ++cy) {
            int yi = y0 + cy;
            if ((unsigned)yi < (unsigned)HH) {
                float wyc = cy ? wy : (1.0f - wy);
#pragma unroll
                for (int cx = 0; cx < 2; ++cx) {
                    int xi = x0 + cx;
                    if ((unsigned)xi < (unsigned)WWD) {
                        float wxc = cx ? wx : (1.0f - wx);
                        s += wyc * wxc * vbase[(size_t)(yi * WWD + xi) * CDIM];
                    }
                }
            }
        }
        acc += w[p] * s;
    }
    attn[(size_t)bq * CDIM + h * 32 + lane] = (bf16_t)acc;
}

// ---------------------------------------------------------------------------
// out = LayerNorm(out + y) * g + b  (fp32), also emit bf16 copy for next GEMM.
// One wave per row of 256; shfl_xor tree reduction across 32 lanes (wave32).
// ---------------------------------------------------------------------------
__global__ __launch_bounds__(256)
void add_layernorm(float*       __restrict__ out,
                   const float* __restrict__ y,
                   const float* __restrict__ g,
                   const float* __restrict__ bb,
                   bf16_t*      __restrict__ outb,
                   int M)
{
    int row = blockIdx.x * 8 + (threadIdx.x >> 5);
    if (row >= M) return;
    int lane = threadIdx.x & 31;
    float v[8];
    float s = 0.0f, s2 = 0.0f;
#pragma unroll
    for (int j = 0; j < 8; ++j) {
        int c = lane + j * 32;
        float t = out[(size_t)row * CDIM + c] + y[(size_t)row * CDIM + c];
        v[j] = t; s += t; s2 += t * t;
    }
#pragma unroll
    for (int o = 16; o > 0; o >>= 1) {
        s  += __shfl_xor(s,  o, 32);
        s2 += __shfl_xor(s2, o, 32);
    }
    float mean = s  * (1.0f / 256.0f);
    float var  = s2 * (1.0f / 256.0f) - mean * mean;
    float rstd = rsqrtf(var + 1e-5f);
#pragma unroll
    for (int j = 0; j < 8; ++j) {
        int c = lane + j * 32;
        float o = (v[j] - mean) * rstd * g[c] + bb[c];
        out [(size_t)row * CDIM + c] = o;
        outb[(size_t)row * CDIM + c] = (bf16_t)o;
    }
}

// ---------------------------------------------------------------------------
// Sine positional encoding (mmdet, zero mask): pos[hw, c], channels 0..127
// from (qy+1), 128..255 from (qx+1).
// ---------------------------------------------------------------------------
__global__ void sine_pos(float* __restrict__ pos)
{
    int idx = blockIdx.x * 256 + threadIdx.x;      // HWSZ*256 total
    int hw  = idx >> 8;
    int c   = idx & 255;
    int qx  = hw % WWD, qy = hw / WWD;
    int j   = c & 127;
    float coord = (c < 128) ? (float)(qy + 1) : (float)(qx + 1);
    float expo  = (float)(2 * (j >> 1)) * (1.0f / 128.0f);
    float freq  = __powf(10000.0f, expo);
    float ph    = coord / freq;
    pos[idx] = (j & 1) ? __cosf(ph) : __sinf(ph);
}

// q = bf16(out + pos)  (pos broadcast over batch)
__global__ void q_add_pos(const float* __restrict__ out,
                          const float* __restrict__ pos,
                          bf16_t*      __restrict__ q)
{
    int idx = blockIdx.x * 256 + threadIdx.x;      // MTOT*256 total
    int m   = idx >> 8;
    int c   = idx & 255;
    int hw  = m % HWSZ;
    q[idx] = (bf16_t)(out[idx] + pos[(size_t)hw * CDIM + c]);
}

__global__ void cast_to_bf16(const float* __restrict__ s, bf16_t* __restrict__ d, int n)
{
    int i = blockIdx.x * 256 + threadIdx.x;
    if (i < n) d[i] = (bf16_t)s[i];
}

__global__ void zero_bf16(bf16_t* __restrict__ d, int n)
{
    int i = blockIdx.x * 256 + threadIdx.x;
    if (i < n) d[i] = (bf16_t)0.0f;
}

// bev [B,C,HW] fp32 -> x [B,HW,C] bf16 (tiled transpose)
__global__ void transpose_chw_to_hwc(const float* __restrict__ in, bf16_t* __restrict__ outp)
{
    __shared__ float tile[32][33];
    int b   = blockIdx.z;
    int hw0 = blockIdx.x * 32, c0 = blockIdx.y * 32;
    int tx  = threadIdx.x,     ty = threadIdx.y;
    for (int j = ty; j < 32; j += 8)
        tile[j][tx] = in[((size_t)b * CDIM + c0 + j) * HWSZ + hw0 + tx];
    __syncthreads();
    for (int j = ty; j < 32; j += 8)
        outp[((size_t)(b * HWSZ + hw0 + j)) * CDIM + c0 + tx] = (bf16_t)tile[tx][j];
}

// out [B,HW,C] fp32 -> d_out [B,C,HW] fp32
__global__ void transpose_hwc_to_chw(const float* __restrict__ in, float* __restrict__ outp)
{
    __shared__ float tile[32][33];
    int b   = blockIdx.z;
    int hw0 = blockIdx.x * 32, c0 = blockIdx.y * 32;
    int tx  = threadIdx.x,     ty = threadIdx.y;
    for (int j = ty; j < 32; j += 8)
        tile[j][tx] = in[((size_t)(b * HWSZ + hw0 + j)) * CDIM + c0 + tx];
    __syncthreads();
    for (int j = ty; j < 32; j += 8)
        outp[((size_t)b * CDIM + c0 + j) * HWSZ + hw0 + tx] = tile[j][tx];
}

// ---------------------------------------------------------------------------
// Host orchestration
// ---------------------------------------------------------------------------
extern "C" void kernel_launch(void* const* d_in, const int* in_sizes, int n_in,
                              void* d_out, int out_size, void* d_ws, size_t ws_size,
                              hipStream_t stream)
{
    (void)in_sizes; (void)n_in; (void)out_size; (void)ws_size;

    const float* bev  = (const float*)d_in[0];
    const float* qw   = (const float*)d_in[1];
    const float* qb   = (const float*)d_in[2];
    const float* offw = (const float*)d_in[3];
    const float* offb = (const float*)d_in[4];
    const float* aww  = (const float*)d_in[5];
    const float* awb  = (const float*)d_in[6];
    const float* vpw  = (const float*)d_in[7];
    const float* vpb  = (const float*)d_in[8];
    const float* opw  = (const float*)d_in[9];
    const float* opb  = (const float*)d_in[10];
    const float* ln1g = (const float*)d_in[11];
    const float* ln1b = (const float*)d_in[12];
    const float* l1w  = (const float*)d_in[13];
    const float* l1b  = (const float*)d_in[14];
    const float* l2w  = (const float*)d_in[15];
    const float* l2b  = (const float*)d_in[16];
    const float* ln2g = (const float*)d_in[17];
    const float* ln2b = (const float*)d_in[18];

    char* ws = (char*)d_ws;
    float*  out_f32 = (float*) (ws);                    // M*256 f32   (running residual)
    float*  fbuf    = (float*) (ws + 81920000ULL);      // M*256 f32   (value / gemm-out share)
    bf16_t* out_b   = (bf16_t*)(ws + 163840000ULL);     // M*256 bf16  (bf16 mirror of out)
    float*  pos     = (float*) (ws + 204800000ULL);     // HW*256 f32
    bf16_t* b16buf  = (bf16_t*)(ws + 245760000ULL);     // M*256 bf16  (x / q / attn / ffn-h share)
    float*  offbuf  = (float*) (ws + 286720000ULL);     // M*64  f32
    float*  awbuf   = (float*) (ws + 307200000ULL);     // M*32  f32
    bf16_t* wts     = (bf16_t*)(ws + 317440000ULL);     // bf16 weights (~2 MB, short-N padded)

    // --- weight bf16 staging (off/aw padded to 128 rows for unguarded loads) ---
    bf16_t* wq = wts;
    size_t  o  = 65536;
    bf16_t *wvp[3], *woff[3], *waw[3], *wop[3], *wl1[3], *wl2[3];
    for (int l = 0; l < 3; ++l) {
        wvp[l]  = wts + o; o += 65536;
        woff[l] = wts + o; o += 128 * 256;   // 64 valid + 64 zero rows
        waw[l]  = wts + o; o += 128 * 256;   // 32 valid + 96 zero rows
        wop[l]  = wts + o; o += 65536;
        wl1[l]  = wts + o; o += 65536;
        wl2[l]  = wts + o; o += 65536;
    }
    cast_to_bf16<<<256, 256, 0, stream>>>(qw, wq, 65536);
    for (int l = 0; l < 3; ++l) {
        cast_to_bf16<<<256, 256, 0, stream>>>(vpw + (size_t)l * 65536, wvp[l], 65536);
        cast_to_bf16<<<64,  256, 0, stream>>>(offw + (size_t)l * 64 * 256, woff[l], 64 * 256);
        zero_bf16   <<<64,  256, 0, stream>>>(woff[l] + 64 * 256, 64 * 256);
        cast_to_bf16<<<32,  256, 0, stream>>>(aww + (size_t)l * 32 * 256, waw[l], 32 * 256);
        zero_bf16   <<<96,  256, 0, stream>>>(waw[l] + 32 * 256, 96 * 256);
        cast_to_bf16<<<256, 256, 0, stream>>>(opw + (size_t)l * 65536, wop[l], 65536);
        cast_to_bf16<<<256, 256, 0, stream>>>(l1w + (size_t)l * 65536, wl1[l], 65536);
        cast_to_bf16<<<256, 256, 0, stream>>>(l2w + (size_t)l * 65536, wl2[l], 65536);
    }

    // --- input transpose + positional encoding ---
    transpose_chw_to_hwc<<<dim3(HWSZ / 32, CDIM / 32, BSZ), dim3(32, 8), 0, stream>>>(bev, b16buf);
    sine_pos<<<HWSZ, 256, 0, stream>>>(pos);

    const dim3 gB(256);
    const dim3 g256(CDIM / 128, MTOT / 128);   // (2, 625)
    const dim3 gN1(1, MTOT / 128);

    // out = x @ qw^T + qb   (fp32 + bf16 mirror)
    gemm_bf16_wmma<256, false, true, true><<<g256, gB, 0, stream>>>(
        b16buf, wq, qb, out_f32, out_b, MTOT, 256);

    for (int l = 0; l < 3; ++l) {
        // q = bf16(out + pos)
        q_add_pos<<<MTOT, 256, 0, stream>>>(out_f32, pos, b16buf);
        // value = out @ vp^T + b
        gemm_bf16_wmma<256, false, true, false><<<g256, gB, 0, stream>>>(
            out_b, wvp[l], vpb + (size_t)l * 256, fbuf, (bf16_t*)nullptr, MTOT, 256);
        // offsets / attention logits from q
        gemm_bf16_wmma<256, false, true, false><<<gN1, gB, 0, stream>>>(
            b16buf, woff[l], offb + (size_t)l * 64, offbuf, (bf16_t*)nullptr, MTOT, 64);
        gemm_bf16_wmma<256, false, true, false><<<gN1, gB, 0, stream>>>(
            b16buf, waw[l], awb + (size_t)l * 32, awbuf, (bf16_t*)nullptr, MTOT, 32);
        // deformable sampling -> attn (bf16), reuses b16buf
        dfa_sample<<<MTOT, 256, 0, stream>>>(fbuf, offbuf, awbuf, b16buf);
        // y = attn @ op^T + b  (reuses fbuf)
        gemm_bf16_wmma<256, false, true, false><<<g256, gB, 0, stream>>>(
            b16buf, wop[l], opb + (size_t)l * 256, fbuf, (bf16_t*)nullptr, MTOT, 256);
        // out = LN(out + y)
        add_layernorm<<<MTOT / 8, 256, 0, stream>>>(
            out_f32, fbuf, ln1g + (size_t)l * 256, ln1b + (size_t)l * 256, out_b, MTOT);
        // FFN: h = relu(out @ l1^T + b) (bf16, reuses b16buf)
        gemm_bf16_wmma<256, true, false, true><<<g256, gB, 0, stream>>>(
            out_b, wl1[l], l1b + (size_t)l * 256, (float*)nullptr, b16buf, MTOT, 256);
        // y2 = h @ l2^T + b  (reuses fbuf)
        gemm_bf16_wmma<256, false, true, false><<<g256, gB, 0, stream>>>(
            b16buf, wl2[l], l2b + (size_t)l * 256, fbuf, (bf16_t*)nullptr, MTOT, 256);
        // out = LN(out + y2)
        add_layernorm<<<MTOT / 8, 256, 0, stream>>>(
            out_f32, fbuf, ln2g + (size_t)l * 256, ln2b + (size_t)l * 256, out_b, MTOT);
    }

    // final [B,HW,C] -> [B,C,H,W]
    transpose_hwc_to_chw<<<dim3(HWSZ / 32, CDIM / 32, BSZ), dim3(32, 8), 0, stream>>>(
        out_f32, (float*)d_out);
}
